// MultiGranularityAdaptiveAlignment_1726576853501
// MI455X (gfx1250) — matrix operand order repaired and checked by reading, not compiled, not run on hardware
//
#include <hip/hip_runtime.h>
#include <hip/hip_bf16.h>

// ---------------- types ----------------
typedef __attribute__((ext_vector_type(16))) __bf16    v16bf;
typedef __attribute__((ext_vector_type(8)))  float     v8f;
typedef __attribute__((ext_vector_type(4)))  unsigned  v4u;
typedef __attribute__((ext_vector_type(8)))  int       v8i;
typedef __attribute__((ext_vector_type(4)))  int       v4i;

union Frag { unsigned u[8]; v16bf v; };

#if defined(__has_builtin)
#if __has_builtin(__builtin_amdgcn_tensor_load_to_lds) && __has_builtin(__builtin_amdgcn_s_wait_tensorcnt)
#define USE_TDM 1
#endif
#endif

__device__ __forceinline__ unsigned short f2bf(float f) {
  unsigned u = __float_as_uint(f);
  unsigned rnd = ((u >> 16) & 1u) + 0x7FFFu;   // round-to-nearest-even
  return (unsigned short)((u + rnd) >> 16);
}

__device__ __forceinline__ float softplusf(float x) {
  return (x > 20.f) ? x : log1pf(expf(x));
}

__device__ __forceinline__ unsigned lds_off(const void* p) {
  // low 32 bits of a generic pointer into LDS == byte offset from LDS_BASE
  return (unsigned)(unsigned long long)p;
}

#ifdef USE_TDM
// 2-D tile DMA: global (tensor_d0 x tensor_d1, dim0-stride in elements of 2B)
// -> LDS contiguous [tile_d1][tile_d0] bf16.  D# per CDNA5 ISA 8.3/8.4:
// group0 = {count=1, lds_addr, global_addr, type=2}, group1 = dims/strides.
// 6-arg builtin: (g0 v4u, g1 v8i, g2 v4i, g3 v4i, extra v8i, cpol).
__device__ __forceinline__ void tdm_load_2d(unsigned lds_addr, const void* gptr,
                                            unsigned tensor_d0, unsigned tensor_d1,
                                            unsigned tile_d0, unsigned tile_d1,
                                            unsigned long long stride0_elems) {
  unsigned long long ga = (unsigned long long)gptr;
  v4u g0;
  g0.x = 1u;                                   // count=1, user descriptor
  g0.y = lds_addr;                             // LDS byte address
  g0.z = (unsigned)ga;                         // global addr [31:0]
  g0.w = (unsigned)((ga >> 32) & 0x1FFFFFFu) | (2u << 30);  // addr[56:32] | type=2
  v8i g1;
  g1[0] = 0x10000;                                                  // data_size=2B
  g1[1] = (int)((tensor_d0 & 0xFFFFu) << 16);                       // tensor_dim0[15:0]
  g1[2] = (int)((tensor_d0 >> 16) | ((tensor_d1 & 0xFFFFu) << 16)); // dim0 hi | dim1 lo
  g1[3] = (int)(((tensor_d1 >> 16) & 0xFFFFu) | (tile_d0 << 16));   // dim1 hi | tile_dim0
  g1[4] = (int)(tile_d1 & 0xFFFFu);                                 // tile_dim1 (tile_dim2=0)
  g1[5] = (int)(stride0_elems & 0xFFFFFFFFu);                       // dim0_stride lo
  g1[6] = (int)((stride0_elems >> 32) & 0xFFFFu);                   // dim0_stride hi
  g1[7] = 0;
  v4i z4; z4[0] = 0; z4[1] = 0; z4[2] = 0; z4[3] = 0;
  v8i z8; for (int i = 0; i < 8; ++i) z8[i] = 0;
  __builtin_amdgcn_tensor_load_to_lds(g0, g1, z4, z4, z8, 0);
}
#endif

// ---------------- concat + convert to padded NHWC bf16 ----------------
// out: [B][82][82][512] bf16, zero halo of 1 pixel
__global__ void concat_pad_nhwc_kernel(const float* __restrict__ a,
                                       const float* __restrict__ b,
                                       unsigned short* __restrict__ out) {
  int i = blockIdx.x * 256 + threadIdx.x;
  const int total = 2 * 82 * 82 * 512;
  if (i >= total) return;
  int c  = i & 511;
  int xp = (i >> 9) % 82;
  int yp = (i / (512 * 82)) % 82;
  int bb = i / (512 * 82 * 82);
  unsigned short v = 0;
  if (xp >= 1 && xp <= 80 && yp >= 1 && yp <= 80) {
    int x = xp - 1, y = yp - 1;
    float f = (c < 256) ? a[(size_t)(bb * 256 + c) * 6400 + y * 80 + x]
                        : b[(size_t)(bb * 256 + (c - 256)) * 6400 + y * 80 + x];
    v = f2bf(f);
  }
  out[i] = v;
}

__global__ void fill_zero_u16_kernel(unsigned short* __restrict__ p, int n) {
  int i = blockIdx.x * 256 + threadIdx.x;
  if (i < n) p[i] = 0;
}

// pack conv weight [Cout][Cin][3][3] f32 -> [9][Cin/64][Cout_pad][2][32] bf16
// (per (tap, 64k chunk-pair): each 16-row wave slice is contiguous)
__global__ void pack_conv_weight_kernel(const float* __restrict__ w,
                                        unsigned short* __restrict__ dst,
                                        int Cin, int Cout, int Cout_pad) {
  int i = blockIdx.x * 256 + threadIdx.x;
  int nch2 = Cin >> 6;
  int total = 9 * nch2 * Cout_pad * 64;
  if (i >= total) return;
  int c   = i & 31;
  int q   = (i >> 5) & 1;
  int m   = (i >> 6) % Cout_pad;
  int blk = i / (Cout_pad * 64);
  int chp = blk % nch2;
  int t   = blk / nch2;
  float v = (m < Cout) ? w[(size_t)(m * Cin + (chp * 2 + q) * 32 + c) * 9 + t] : 0.f;
  dst[i] = f2bf(v);
}

// pack dcn weight [256][32][3][3] f32 -> [9][8 g][32 m][32 c] bf16
__global__ void pack_dcn_weight_kernel(const float* __restrict__ w,
                                       unsigned short* __restrict__ dst) {
  int i = blockIdx.x * 256 + threadIdx.x;
  if (i >= 9 * 8 * 32 * 32) return;
  int c = i & 31;
  int m = (i >> 5) & 31;
  int g = (i >> 10) & 7;
  int t = i >> 13;
  dst[i] = f2bf(w[(size_t)((g * 32 + m) * 32 + c) * 9 + t]);
}

// ---------------- global average pool ----------------
__global__ void pool_kernel(const float* __restrict__ rgb,
                            const float* __restrict__ th,
                            float* __restrict__ xp) {
  int bc = blockIdx.x;            // b*512 + ch
  int b = bc >> 9, ch = bc & 511;
  const float* src = (ch < 256) ? (rgb + (size_t)(b * 256 + ch) * 6400)
                                : (th  + (size_t)(b * 256 + (ch - 256)) * 6400);
  float s = 0.f;
  for (int i = threadIdx.x; i < 6400; i += 64) s += src[i];
  __shared__ float red[64];
  red[threadIdx.x] = s;
  __syncthreads();
  for (int off = 32; off > 0; off >>= 1) {
    if (threadIdx.x < off) red[threadIdx.x] += red[threadIdx.x + off];
    __syncthreads();
  }
  if (threadIdx.x == 0) xp[bc] = red[0] * (1.0f / 6400.0f);
}

// ---------------- tiny MLPs for global / channel gates ----------------
__global__ void mlp_kernel(const float* __restrict__ xp,
                           const float* __restrict__ gw1, const float* __restrict__ gb1,
                           const float* __restrict__ gw2, const float* __restrict__ gb2,
                           const float* __restrict__ cw1, const float* __restrict__ cb1,
                           const float* __restrict__ cw2, const float* __restrict__ cb2,
                           float* __restrict__ gout, float* __restrict__ cout_) {
  int b = blockIdx.x;
  int t = threadIdx.x;            // 256 threads
  __shared__ float sx[512];
  __shared__ float h1g[32];
  __shared__ float h1c[256];
  sx[t] = xp[b * 512 + t];
  sx[t + 256] = xp[b * 512 + 256 + t];
  __syncthreads();
  if (t < 32) {
    float s = gb1[t];
    for (int i = 0; i < 512; ++i) s += gw1[t * 512 + i] * sx[i];
    h1g[t] = fmaxf(s, 0.f);
  }
  {
    float s = cb1[t];
    for (int i = 0; i < 512; ++i) s += cw1[t * 512 + i] * sx[i];
    h1c[t] = fmaxf(s, 0.f);
  }
  __syncthreads();
  if (t == 0) {
    float s = gb2[0];
    for (int i = 0; i < 32; ++i) s += gw2[i] * h1g[i];
    gout[b] = softplusf(s);
  }
  if (t < 8) {
    float s = cb2[t];
    for (int i = 0; i < 256; ++i) s += cw2[t * 256 + i] * h1c[i];
    cout_[b * 8 + t] = softplusf(s);
  }
}

// ---------------- offsets = base * (softmax-weighted gates) ----------------
__global__ void combine_kernel(float* __restrict__ base,
                               const float* __restrict__ g,
                               const float* __restrict__ c,
                               const float* __restrict__ s,
                               const float* __restrict__ wg,
                               const float* __restrict__ wc,
                               const float* __restrict__ ws) {
  int i = blockIdx.x * 256 + threadIdx.x;
  const int total = 2 * 144 * 6400;
  if (i >= total) return;
  float e0 = expf(wg[0]), e1 = expf(wc[0]), e2 = expf(ws[0]);
  float inv = 1.f / (e0 + e1 + e2);
  int pix = i % 6400;
  int ch  = (i / 6400) % 144;
  int b   = i / (144 * 6400);
  float comb = (e0 * g[b] + e1 * c[b * 8 + (ch % 72) / 9] + e2 * s[b * 6400 + pix]) * inv;
  base[i] = base[i] * comb;
}

// ---------------- implicit-GEMM 3x3 conv via WMMA bf16 + TDM ----------------
// Input: padded NHWC bf16 [B][82][82][Cin]. Weights pre-packed as
// [9][Cin/64][Cout_pad][2][32] bf16. 128 thr = 4 waves; WG computes 16 px x
// 64 couts. Per 64-k step EVERY wave issues TENSOR_LOAD_TO_LDS for its own
// disjoint slice (its 16-cout weight rows: contiguous 2KB; 4 of the 16
// im2col pixels: 2-D 64ch x 4px tile), waits its TENSORcnt, barriers, then
// runs 2 WMMAs. Uniform control flow, no bounds checks.
__global__ __launch_bounds__(128)
void conv3x3_wmma_kernel(const unsigned short* __restrict__ in, int Cin,
                         const unsigned short* __restrict__ wp,
                         const float* __restrict__ bias,
                         int Cout, int Cout_pad, int act, int outMode,
                         void* __restrict__ outp) {
  const int H = 80, W = 80, HW = 6400, P = 82;
  int x0 = blockIdx.x * 16;
  int y  = blockIdx.y;
  int mgroups = Cout_pad >> 6;
  int b  = blockIdx.z / mgroups;
  int mg = blockIdx.z % mgroups;
  int tid = threadIdx.x, wave = tid >> 5, lane = tid & 31;

  __shared__ unsigned short sA[64 * 64];    // [m][q*32 + c]
  __shared__ unsigned short sB[16 * 64];    // [n][q*32 + c]

  v8f acc;
  #pragma unroll
  for (int v = 0; v < 8; ++v) acc[v] = 0.f;

  int nch2 = Cin >> 6;
#ifdef USE_TDM
  unsigned ldsA = lds_off(&sA[0]) + (unsigned)wave * 2048;  // 16 rows x 64 x 2B
  unsigned ldsB = lds_off(&sB[0]) + (unsigned)wave * 512;   // 4 px  x 64 x 2B
#endif
  int half = lane >> 4, mr = lane & 15;

  for (int t = 0; t < 9; ++t) {
    int yy  = y + t / 3;        // padded coords
    int xx0 = x0 + t % 3;
    const unsigned short* pB0 = in + ((size_t)(b * P + yy) * P + xx0) * Cin;
    for (int chp = 0; chp < nch2; ++chp) {
#ifdef USE_TDM
      // every wave: its 4 pixels of the im2col tile (64 ch x 4 px, px stride Cin)
      tdm_load_2d(ldsB, pB0 + chp * 64 + (size_t)(wave * 4) * Cin,
                  64, 4, 64, 4, (unsigned long long)Cin);
      // every wave: its 16 weight rows (contiguous 1024 elements)
      const unsigned short* pA =
          wp + ((size_t)(t * nch2 + chp) * Cout_pad + mg * 64 + wave * 16) * 64;
      tdm_load_2d(ldsA, pA, 1024, 1, 1024, 1, 1024ull);
      __builtin_amdgcn_s_wait_tensorcnt(0);
#else
      for (int e = tid; e < 1024; e += 128) {      // B: 16 px x 64 ch
        int n = e >> 6, cc = e & 63;
        sB[e] = pB0[(size_t)n * Cin + chp * 64 + cc];
      }
      for (int e = tid; e < 4096; e += 128) {      // A: [64][64]
        sA[e] = wp[((size_t)(t * nch2 + chp) * Cout_pad + mg * 64) * 64 + e];
      }
#endif
      __syncthreads();
      #pragma unroll
      for (int q = 0; q < 2; ++q) {
        Frag a, fb;
        #pragma unroll
        for (int v = 0; v < 8; ++v) {
          int ka = 2 * v + half * 16;
          a.u[v]  = *(const unsigned*)&sA[(wave * 16 + mr) * 64 + q * 32 + ka];
          int kb = 2 * v + half * 16;
          fb.u[v] = *(const unsigned*)&sB[mr * 64 + q * 32 + kb];
        }
        acc = __builtin_amdgcn_wmma_f32_16x16x32_bf16(
            false, a.v, false, fb.v, (short)0, acc, false, false);
      }
      __syncthreads();
    }
  }

  // epilogue: bias + activation
  int n = lane & 15, halfm = lane >> 4;
  int m0 = mg * 64 + wave * 16;
  #pragma unroll
  for (int v = 0; v < 8; ++v) {
    int m = m0 + halfm * 8 + v;
    if (m >= Cout) continue;
    float val = acc[v] + bias[m];
    if (act == 1) val = fmaxf(val, 0.f);
    else if (act == 2) val = tanhf(val);
    else if (act == 3) val = 1.f / (1.f + expf(-val));
    else if (act == 4) val = softplusf(val);
    if (outMode == 1) {  // bf16, padded NHWC
      ((unsigned short*)outp)[((size_t)(b * P + y + 1) * P + (x0 + n + 1)) * Cout + m] = f2bf(val);
    } else {             // f32, NCHW
      ((float*)outp)[(size_t)(b * Cout + m) * HW + y * W + x0 + n] = val;
    }
  }
}

// ---------------- modulated deformable conv (DCNv2) via WMMA ----------------
// One WG (64 thr = 2 waves) per (b, group, 16-px block). Each wave DMAs its
// own 16-row half of the group weight block per tap (TDM); bilinear gather
// (mask-premultiplied) fills the bf16 B tile; 9 WMMAs per wave, f32 accum.
__global__ __launch_bounds__(64)
void dcn_wmma_kernel(const float* __restrict__ rgb,      // NCHW f32
                     const float* __restrict__ offs,     // [B,144,HW]
                     const float* __restrict__ mask,     // [B,72,HW]
                     const unsigned short* __restrict__ dwp, // [9][8][32][32] bf16
                     const float* __restrict__ db,
                     float* __restrict__ out) {
  const int H = 80, W = 80, HW = 6400, G = 8;
  int x0 = blockIdx.x * 16;
  int y  = blockIdx.y;
  int bg = blockIdx.z;
  int b = bg >> 3, g = bg & 7;
  int tid = threadIdx.x;
  int wave = tid >> 5, lane = tid & 31;

  __shared__ int   sIY0[144], sIY1[144], sIX0[144], sIX1[144];
  __shared__ float sWt[144][4];
  __shared__ unsigned short sA[1024];   // [32 m][32 c]
  __shared__ unsigned short sB[512];    // [16 n][32 c]

  for (int i = tid; i < 144; i += 64) {
    int n = i / 9, t = i % 9;
    int ky = t / 3 - 1, kx = t % 3 - 1;
    int pix = y * W + x0 + n;
    float dy = offs[(size_t)(((b * G + g) * 9 + t) * 2 + 0) * HW + pix];
    float dx = offs[(size_t)(((b * G + g) * 9 + t) * 2 + 1) * HW + pix];
    float mm = mask[(size_t)((b * G + g) * 9 + t) * HW + pix];
    float py = dy + (float)(ky + y);
    float px = dx + (float)(kx + x0 + n);
    float fy = floorf(py), fx = floorf(px);
    float ly = py - fy, lx = px - fx;
    int iy = (int)fy, ix = (int)fx;
    bool vy0 = (iy >= 0) && (iy < H);
    bool vy1 = (iy + 1 >= 0) && (iy + 1 < H);
    bool vx0 = (ix >= 0) && (ix < W);
    bool vx1 = (ix + 1 >= 0) && (ix + 1 < W);
    sIY0[i] = min(max(iy, 0), H - 1);
    sIY1[i] = min(max(iy + 1, 0), H - 1);
    sIX0[i] = min(max(ix, 0), W - 1);
    sIX1[i] = min(max(ix + 1, 0), W - 1);
    sWt[i][0] = (1.f - ly) * (1.f - lx) * ((vy0 && vx0) ? mm : 0.f);
    sWt[i][1] = (1.f - ly) * lx         * ((vy0 && vx1) ? mm : 0.f);
    sWt[i][2] = ly * (1.f - lx)         * ((vy1 && vx0) ? mm : 0.f);
    sWt[i][3] = ly * lx                 * ((vy1 && vx1) ? mm : 0.f);
  }
  __syncthreads();

  v8f acc;
  #pragma unroll
  for (int v = 0; v < 8; ++v) acc[v] = 0.f;

  const float* xg = rgb + (size_t)(b * 256 + g * 32) * HW;
#ifdef USE_TDM
  unsigned ldsA = lds_off(&sA[0]) + (unsigned)wave * 1024;  // 16 rows x 32 x 2B
#endif
  int half = lane >> 4, mr = lane & 15;

  for (int t = 0; t < 9; ++t) {
#ifdef USE_TDM
    // each wave: its contiguous 16-row half of the [32][32] weight block
    tdm_load_2d(ldsA, dwp + (size_t)(t * 8 + g) * 1024 + wave * 512,
                512, 1, 512, 1, 512ull);
#else
    for (int e = tid; e < 1024; e += 64) sA[e] = dwp[(size_t)(t * 8 + g) * 1024 + e];
#endif
    for (int e = tid; e < 512; e += 64) {
      int c = e >> 4, n = e & 15;
      int i = n * 9 + t;
      const float* xc = xg + (size_t)c * HW;
      float v = sWt[i][0] * xc[sIY0[i] * W + sIX0[i]]
              + sWt[i][1] * xc[sIY0[i] * W + sIX1[i]]
              + sWt[i][2] * xc[sIY1[i] * W + sIX0[i]]
              + sWt[i][3] * xc[sIY1[i] * W + sIX1[i]];
      sB[n * 32 + c] = f2bf(v);
    }
#ifdef USE_TDM
    __builtin_amdgcn_s_wait_tensorcnt(0);
#endif
    __syncthreads();
    Frag a, fb;
    #pragma unroll
    for (int v = 0; v < 8; ++v) {
      int ka = 2 * v + half * 16;
      a.u[v]  = *(const unsigned*)&sA[(wave * 16 + mr) * 32 + ka];
      int kb = 2 * v + half * 16;
      fb.u[v] = *(const unsigned*)&sB[mr * 32 + kb];
    }
    acc = __builtin_amdgcn_wmma_f32_16x16x32_bf16(
        false, a.v, false, fb.v, (short)0, acc, false, false);
    __syncthreads();
  }

  int n = lane & 15, halfm = lane >> 4;
  #pragma unroll
  for (int v = 0; v < 8; ++v) {
    int m = halfm * 8 + v;
    int cout = g * 32 + wave * 16 + m;
    out[(size_t)(b * 256 + cout) * HW + y * W + x0 + n] = acc[v] + db[cout];
  }
}

// ---------------- launch ----------------
extern "C" void kernel_launch(void* const* d_in, const int* in_sizes, int n_in,
                              void* d_out, int out_size, void* d_ws, size_t ws_size,
                              hipStream_t stream) {
  const int B = 2, H = 80, W = 80, HW = H * W, C = 256, G = 8, HID = 64, P = 82;
  (void)in_sizes; (void)n_in; (void)out_size; (void)ws_size;

  const float* rgb = (const float*)d_in[0];
  const float* th  = (const float*)d_in[1];
  const float* ow1 = (const float*)d_in[2];  const float* ob1 = (const float*)d_in[3];
  const float* ow2 = (const float*)d_in[4];  const float* ob2 = (const float*)d_in[5];
  const float* ow3 = (const float*)d_in[6];  const float* ob3 = (const float*)d_in[7];
  const float* gw1 = (const float*)d_in[8];  const float* gb1 = (const float*)d_in[9];
  const float* gw2 = (const float*)d_in[10]; const float* gb2 = (const float*)d_in[11];
  const float* cw1 = (const float*)d_in[12]; const float* cb1 = (const float*)d_in[13];
  const float* cw2 = (const float*)d_in[14]; const float* cb2 = (const float*)d_in[15];
  const float* sw1 = (const float*)d_in[16]; const float* sb1 = (const float*)d_in[17];
  const float* sw2 = (const float*)d_in[18]; const float* sb2 = (const float*)d_in[19];
  const float* mw1 = (const float*)d_in[20]; const float* mb1 = (const float*)d_in[21];
  const float* mw2 = (const float*)d_in[22]; const float* mb2 = (const float*)d_in[23];
  const float* wgl = (const float*)d_in[24];
  const float* wch = (const float*)d_in[25];
  const float* wsp = (const float*)d_in[26];
  const float* dw  = (const float*)d_in[27]; const float* db  = (const float*)d_in[28];

  size_t off = 0;
  auto alloc = [&](size_t bytes) -> void* {
    void* r = (char*)d_ws + off;
    off += (bytes + 255) & ~(size_t)255;
    return r;
  };
  unsigned short* xbf  = (unsigned short*)alloc((size_t)B * P * P * 512 * 2);
  unsigned short* h1   = (unsigned short*)alloc((size_t)B * P * P * HID * 2);
  unsigned short* h2   = (unsigned short*)alloc((size_t)B * P * P * HID * 2);
  float* baseoff       = (float*)alloc((size_t)B * 144 * HW * 4);
  float* sbuf          = (float*)alloc((size_t)B * HW * 4);
  float* maskbuf       = (float*)alloc((size_t)B * 72 * HW * 4);
  float* xp            = (float*)alloc((size_t)B * 512 * 4);
  float* gbuf          = (float*)alloc(16 * 4);
  float* cbuf          = (float*)alloc((size_t)B * 8 * 4);
  unsigned short* ow1p = (unsigned short*)alloc((size_t)9 * 8 * 64  * 64 * 2);
  unsigned short* ow2p = (unsigned short*)alloc((size_t)9 * 1 * 64  * 64 * 2);
  unsigned short* ow3p = (unsigned short*)alloc((size_t)9 * 1 * 192 * 64 * 2);
  unsigned short* sw1p = (unsigned short*)alloc((size_t)9 * 8 * 64  * 64 * 2);
  unsigned short* sw2p = (unsigned short*)alloc((size_t)9 * 1 * 64  * 64 * 2);
  unsigned short* mw1p = (unsigned short*)alloc((size_t)9 * 8 * 64  * 64 * 2);
  unsigned short* mw2p = (unsigned short*)alloc((size_t)9 * 1 * 128 * 64 * 2);
  unsigned short* dwp  = (unsigned short*)alloc((size_t)9 * 8 * 32  * 32 * 2);

  auto gsz = [](int n) { return (n + 255) / 256; };

  // layout transforms
  {
    int n = B * P * P * 512;
    concat_pad_nhwc_kernel<<<gsz(n), 256, 0, stream>>>(rgb, th, xbf);
  }
  fill_zero_u16_kernel<<<gsz(B * P * P * HID), 256, 0, stream>>>(h1, B * P * P * HID);
  fill_zero_u16_kernel<<<gsz(B * P * P * HID), 256, 0, stream>>>(h2, B * P * P * HID);

  auto pack = [&](const float* src, unsigned short* dst, int Cin, int Cout, int Cpad) {
    int n = 9 * (Cin >> 6) * Cpad * 64;
    pack_conv_weight_kernel<<<gsz(n), 256, 0, stream>>>(src, dst, Cin, Cout, Cpad);
  };
  pack(ow1, ow1p, 512, HID, 64);
  pack(ow2, ow2p, 64,  HID, 64);
  pack(ow3, ow3p, 64,  144, 192);
  pack(sw1, sw1p, 512, HID, 64);
  pack(sw2, sw2p, 64,  1,   64);
  pack(mw1, mw1p, 512, HID, 64);
  pack(mw2, mw2p, 64,  72,  128);
  pack_dcn_weight_kernel<<<gsz(9 * 8 * 32 * 32), 256, 0, stream>>>(dw, dwp);

  // global / channel gates
  pool_kernel<<<B * 512, 64, 0, stream>>>(rgb, th, xp);
  mlp_kernel<<<B, 256, 0, stream>>>(xp, gw1, gb1, gw2, gb2, cw1, cb1, cw2, cb2,
                                    gbuf, cbuf);

  const dim3 cblk(128);
  // offset net: 512 -> 64 (relu) -> 64 (relu) -> 144 (tanh)
  conv3x3_wmma_kernel<<<dim3(5, 80, B * 1), cblk, 0, stream>>>(xbf, 512, ow1p, ob1, HID, 64,  1, 1, h1);
  conv3x3_wmma_kernel<<<dim3(5, 80, B * 1), cblk, 0, stream>>>(h1,  64,  ow2p, ob2, HID, 64,  1, 1, h2);
  conv3x3_wmma_kernel<<<dim3(5, 80, B * 3), cblk, 0, stream>>>(h2,  64,  ow3p, ob3, 144, 192, 2, 0, baseoff);
  // spatial net: 512 -> 64 (relu) -> 1 (softplus)   (reuses h1)
  conv3x3_wmma_kernel<<<dim3(5, 80, B * 1), cblk, 0, stream>>>(xbf, 512, sw1p, sb1, HID, 64,  1, 1, h1);
  conv3x3_wmma_kernel<<<dim3(5, 80, B * 1), cblk, 0, stream>>>(h1,  64,  sw2p, sb2, 1,   64,  4, 0, sbuf);
  // offsets = tanh(base) * softmax-weighted gates
  {
    int n = B * 144 * HW;
    combine_kernel<<<gsz(n), 256, 0, stream>>>(baseoff, gbuf, cbuf, sbuf, wgl, wch, wsp);
  }
  // mask net: 512 -> 64 (relu) -> 72 (sigmoid)   (reuses h1)
  conv3x3_wmma_kernel<<<dim3(5, 80, B * 1), cblk, 0, stream>>>(xbf, 512, mw1p, mb1, HID, 64,  1, 1, h1);
  conv3x3_wmma_kernel<<<dim3(5, 80, B * 2), cblk, 0, stream>>>(h1,  64,  mw2p, mb2, 72,  128, 3, 0, maskbuf);

  // deformable conv
  dcn_wmma_kernel<<<dim3(5, 80, B * G), 64, 0, stream>>>(rgb, baseoff, maskbuf,
                                                         dwp, db, (float*)d_out);
}
